// SequenceLabeling_45973329936855
// MI455X (gfx1250) — matrix operand order repaired and checked by reading, not compiled, not run on hardware
//
#include <hip/hip_runtime.h>
#include <hip/hip_bf16.h>
#include <math.h>

#define BB 64
#define TT 512
#define EE 768
#define HH 128
#define CC 18
#define GG 512            // 4*H
#define MM (BB*TT)        // 32768

typedef __attribute__((ext_vector_type(16))) __bf16         v16bf;
typedef __attribute__((ext_vector_type(8)))  float          v8f;
typedef __attribute__((ext_vector_type(8)))  unsigned short ushort8;
typedef __attribute__((ext_vector_type(16))) unsigned short ushort16;

static __device__ __forceinline__ unsigned short f32_to_bf16(float f) {
  unsigned int u = __builtin_bit_cast(unsigned int, f);
  u += 0x7FFFu + ((u >> 16) & 1u);              // round-to-nearest-even
  return (unsigned short)(u >> 16);
}

static __device__ __forceinline__ v16bf frag16(ushort8 lo, ushort8 hi) {
  ushort16 t = __builtin_shufflevector(lo, hi, 0,1,2,3,4,5,6,7,8,9,10,11,12,13,14,15);
  return __builtin_bit_cast(v16bf, t);
}

static __device__ __forceinline__ float sigf(float x) {
  return 1.0f / (1.0f + __expf(-x));
}

// ---------------------------------------------------------------- conversion
__global__ void cvt_f32_bf16(const float* __restrict__ s,
                             unsigned short* __restrict__ d, long n) {
  long i = (long)blockIdx.x * blockDim.x + threadIdx.x;
  long stride = (long)gridDim.x * blockDim.x;
  for (; i < n; i += stride) d[i] = f32_to_bf16(s[i]);
}

// ------------------------------------------- pre-activation GEMM (both dirs)
// pre[dir][m][g] = sum_k x[m,k]*Wih[dir][g,k] + b[dir][g]
// block = 256 threads (8 waves); block tile 128(M) x 64(N); wave tile 16x64.
__global__ __launch_bounds__(256) void pre_gemm(
    const unsigned short* __restrict__ xb,        // MM x EE bf16, row major
    const unsigned short* __restrict__ wih_bf,    // 2 x GG x EE bf16, row major
    const float* __restrict__ b_f, const float* __restrict__ b_b,
    float* __restrict__ pre) {                    // 2 x MM x GG f32
  const int dir  = blockIdx.z;
  const unsigned short* W = wih_bf + (size_t)dir * GG * EE;
  const float* bias = dir ? b_b : b_f;
  float* preD = pre + (size_t)dir * MM * GG;

  const int lane  = threadIdx.x & 31;
  const int wave  = threadIdx.x >> 5;
  const int mbase = blockIdx.x * 128 + wave * 16;
  const int nbase = blockIdx.y * 64;
  const int koffA = (lane >> 4) * 8;    // ISA 16-bit A layout
  const int koffB = (lane >> 4) * 8;    // ISA 16-bit B layout

  v8f acc[4] = {};
  const unsigned short* arow = xb + (size_t)(mbase + (lane & 15)) * EE;

  for (int k = 0; k < EE; k += 32) {
    ushort8 a0 = *(const ushort8*)(arow + k + koffA);
    ushort8 a1 = *(const ushort8*)(arow + k + koffA + 16);
    v16bf a = frag16(a0, a1);
#pragma unroll
    for (int j = 0; j < 4; j++) {
      const unsigned short* brow =
          W + (size_t)(nbase + j * 16 + (lane & 15)) * EE + k + koffB;
      ushort8 b0 = *(const ushort8*)(brow);
      ushort8 b1 = *(const ushort8*)(brow + 16);
      v16bf b = frag16(b0, b1);
      acc[j] = __builtin_amdgcn_wmma_f32_16x16x32_bf16(
          false, a, false, b, (short)0, acc[j], false, false);
    }
  }
#pragma unroll
  for (int j = 0; j < 4; j++) {
    int col = nbase + j * 16 + (lane & 15);
    float bz = bias[col];
#pragma unroll
    for (int i = 0; i < 8; i++) {
      int row = mbase + i + ((lane >> 4) << 3);
      preD[(size_t)row * GG + col] = acc[j][i] + bz;
    }
  }
}

// ------------------------------------------------------------ LSTM recurrence
// One workgroup (32 waves) per direction; h/c/g live in LDS (320KB WGP LDS).
__global__ __launch_bounds__(1024) void lstm_rec(
    const unsigned short* __restrict__ whh_bf,    // 2 x GG x HH bf16
    const float* __restrict__ pre_all,            // 2 x MM x GG f32
    float* __restrict__ seq) {                    // MM x 2H f32
  const int dir = blockIdx.x;
  const unsigned short* whh = whh_bf + (size_t)dir * GG * HH;
  const float* pre = pre_all + (size_t)dir * MM * GG;

  extern __shared__ char smem[];
  float* g_s = (float*)smem;                                    // 64*512 f32
  float* c_s = (float*)(smem + (size_t)BB * GG * 4);            // 64*128 f32
  unsigned short* h_bf =
      (unsigned short*)(smem + (size_t)BB * GG * 4 + (size_t)BB * HH * 4);

  const int tid  = threadIdx.x;
  const int lane = tid & 31;
  const int wave = tid >> 5;

  for (int i = tid; i < BB * HH; i += 1024) { c_s[i] = 0.0f; h_bf[i] = 0; }
  __syncthreads();

  const int koffA = (lane >> 4) * 8;
  const int koffB = (lane >> 4) * 8;
  const int ncol  = wave * 16 + (lane & 15);       // this wave's N column tile
  const unsigned short* wrow = whh + (size_t)ncol * HH;
  __builtin_prefetch(wrow, 0, 1);                  // global_prefetch_b8

  for (int t = 0; t < TT; t++) {
    // ---- g = h @ Whh^T : 4 M-tiles per wave, K = 128 in 4 WMMA steps
#pragma unroll
    for (int mt = 0; mt < 4; mt++) {
      v8f acc = {};
      const unsigned short* arow = h_bf + (size_t)(mt * 16 + (lane & 15)) * HH;
#pragma unroll
      for (int k = 0; k < HH; k += 32) {
        ushort8 a0 = *(const ushort8*)(arow + k + koffA);
        ushort8 a1 = *(const ushort8*)(arow + k + koffA + 16);
        v16bf a = frag16(a0, a1);
        ushort8 b0 = *(const ushort8*)(wrow + k + koffB);
        ushort8 b1 = *(const ushort8*)(wrow + k + koffB + 16);
        v16bf b = frag16(b0, b1);
        acc = __builtin_amdgcn_wmma_f32_16x16x32_bf16(
            false, a, false, b, (short)0, acc, false, false);
      }
#pragma unroll
      for (int i = 0; i < 8; i++) {
        int row = mt * 16 + i + ((lane >> 4) << 3);
        g_s[row * GG + ncol] = acc[i];
      }
    }
    __syncthreads();

    // ---- gates / state update / store h
    const int tcur = dir ? (TT - 1 - t) : t;
    for (int idx = tid; idx < BB * HH; idx += 1024) {
      int b  = idx >> 7;
      int hc = idx & (HH - 1);
      const float* pr = pre + ((size_t)b * TT + tcur) * GG;
      float iv = g_s[b * GG + hc]          + pr[hc];
      float fv = g_s[b * GG + HH + hc]     + pr[HH + hc];
      float gv = g_s[b * GG + 2 * HH + hc] + pr[2 * HH + hc];
      float ov = g_s[b * GG + 3 * HH + hc] + pr[3 * HH + hc];
      float c = sigf(fv) * c_s[idx] + sigf(iv) * tanhf(gv);
      c_s[idx] = c;
      float h = sigf(ov) * tanhf(c);
      seq[((size_t)b * TT + tcur) * (2 * HH) + dir * HH + hc] = h;
      h_bf[idx] = f32_to_bf16(h);
    }
    __syncthreads();
  }
}

// ------------------------------------------------------------- FC emissions
__global__ __launch_bounds__(32) void fc_kernel(
    const float* __restrict__ seq, const float* __restrict__ Wfc,
    const float* __restrict__ bfc, float* __restrict__ em) {
  int m = blockIdx.x;
  int j = threadIdx.x;
  if (j >= CC) return;
  const float* row = seq + (size_t)m * (2 * HH);
  const float* w   = Wfc + (size_t)j * (2 * HH);
  float s = bfc[j];
  for (int k = 0; k < 2 * HH; k++) s = fmaf(row[k], w[k], s);
  em[(size_t)m * CC + j] = s;
}

__global__ void init_loss(float* out) { out[MM] = 0.0f; }

// ---------------------------------------------- CRF forward + Viterbi (1 wave/seq)
__global__ __launch_bounds__(32) void crf_kernel(
    const float* __restrict__ em, const int* __restrict__ mask,
    const int* __restrict__ labels, const float* __restrict__ startT,
    const float* __restrict__ endT, const float* __restrict__ trans,
    float* __restrict__ out, int* __restrict__ bptr, float invB) {
  const int b = blockIdx.x;
  const int j = threadIdx.x;
  __shared__ float shA[32];
  __shared__ float shV[32];

  const float* emb = em + (size_t)b * TT * CC;
  const int* mb = mask + (size_t)b * TT;
  const int* tb = labels + (size_t)b * TT;

  float trc[CC];
  if (j < CC)
    for (int i = 0; i < CC; i++) trc[i] = trans[i * CC + j];

  float alpha = (j < CC) ? startT[j] + emb[j] : -3.0e38f;
  float vit   = alpha;

  for (int t = 1; t < TT; t++) {
    shA[j] = alpha;
    shV[j] = vit;
    __syncthreads();
    if (j < CC) {
      float m  = -3.0e38f;
      float bv = -3.0e38f;
      int   bi = 0;
      for (int i = 0; i < CC; i++) {
        float ca = shA[i] + trc[i];
        if (ca > m) m = ca;
        float cv = shV[i] + trc[i];
        if (cv > bv) { bv = cv; bi = i; }
      }
      float ss = 0.0f;
      for (int i = 0; i < CC; i++) ss += __expf(shA[i] + trc[i] - m);
      float e  = emb[t * CC + j];
      float na = m + __logf(ss) + e;
      alpha = mb[t] ? na : alpha;
      vit = bv + e;
      bptr[((size_t)b * (TT - 1) + (t - 1)) * CC + j] = bi;
    }
    __syncthreads();
  }

  shA[j] = (j < CC) ? alpha + endT[j] : -3.0e38f;
  shV[j] = (j < CC) ? vit + endT[j] : -3.0e38f;
  __syncthreads();

  if (j == 0) {
    // denominator
    float m = -3.0e38f;
    for (int i = 0; i < CC; i++) m = fmaxf(m, shA[i]);
    float ss = 0.0f;
    for (int i = 0; i < CC; i++) ss += __expf(shA[i] - m);
    float den = m + __logf(ss);
    // numerator
    int tag0 = tb[0];
    float num = startT[tag0] + emb[tag0];
    int nmask = mb[0] ? 1 : 0;
    for (int t = 1; t < TT; t++) {
      float mf = mb[t] ? 1.0f : 0.0f;
      num += (emb[t * CC + tb[t]] + trans[tb[t - 1] * CC + tb[t]]) * mf;
      nmask += mb[t] ? 1 : 0;
    }
    int lastTag = tb[nmask - 1];
    num += endT[lastTag];
    atomicAdd(out + MM, -(num - den) * invB);
    // Viterbi best last + backtrace
    float bv = -3.0e38f; int last = 0;
    for (int i = 0; i < CC; i++) if (shV[i] > bv) { bv = shV[i]; last = i; }
    int tag = last;
    out[(size_t)b * TT + (TT - 1)] = (float)(tag + 1);
    for (int t = TT - 2; t >= 0; t--) {
      tag = bptr[((size_t)b * (TT - 1) + t) * CC + tag];
      out[(size_t)b * TT + t] = (float)(tag + 1);
    }
  }
}

// ---------------------------------------------------------------- launcher
extern "C" void kernel_launch(void* const* d_in, const int* in_sizes, int n_in,
                              void* d_out, int out_size, void* d_ws, size_t ws_size,
                              hipStream_t stream) {
  const float* x      = (const float*)d_in[0];
  const int*   mask   = (const int*)d_in[1];
  const int*   labels = (const int*)d_in[2];
  const float* Wih_f  = (const float*)d_in[3];
  const float* Whh_f  = (const float*)d_in[4];
  const float* b_f    = (const float*)d_in[5];
  const float* Wih_b  = (const float*)d_in[6];
  const float* Whh_b  = (const float*)d_in[7];
  const float* b_b    = (const float*)d_in[8];
  const float* Wfc    = (const float*)d_in[9];
  const float* bfc    = (const float*)d_in[10];
  const float* startT = (const float*)d_in[11];
  const float* endT   = (const float*)d_in[12];
  const float* trans  = (const float*)d_in[13];
  float* out = (float*)d_out;

  char* ws = (char*)d_ws;
  size_t off = 0;
  auto alloc = [&](size_t bytes) -> char* {
    char* p = ws + off;
    off = (off + bytes + 255) & ~(size_t)255;
    return p;
  };
  unsigned short* xb     = (unsigned short*)alloc((size_t)MM * EE * 2);
  unsigned short* wih_bf = (unsigned short*)alloc((size_t)2 * GG * EE * 2);
  unsigned short* whh_bf = (unsigned short*)alloc((size_t)2 * GG * HH * 2);
  float* pre = (float*)alloc((size_t)2 * MM * GG * 4);
  float* seq = (float*)alloc((size_t)MM * 2 * HH * 4);
  float* em  = (float*)alloc((size_t)MM * CC * 4);
  int*   bp  = (int*)alloc((size_t)BB * (TT - 1) * CC * 4);

  // bf16 conversions
  cvt_f32_bf16<<<4096, 256, 0, stream>>>(x, xb, (long)MM * EE);
  cvt_f32_bf16<<<1024, 256, 0, stream>>>(Wih_f, wih_bf, (long)GG * EE);
  cvt_f32_bf16<<<1024, 256, 0, stream>>>(Wih_b, wih_bf + (size_t)GG * EE, (long)GG * EE);
  cvt_f32_bf16<<<256, 256, 0, stream>>>(Whh_f, whh_bf, (long)GG * HH);
  cvt_f32_bf16<<<256, 256, 0, stream>>>(Whh_b, whh_bf + (size_t)GG * HH, (long)GG * HH);

  // pre-activation GEMM: grid (M/128, G/64, 2 dirs)
  pre_gemm<<<dim3(MM / 128, GG / 64, 2), 256, 0, stream>>>(xb, wih_bf, b_f, b_b, pre);

  // recurrence: one WGP-sized workgroup per direction, 180KB dynamic LDS
  size_t lds = (size_t)BB * GG * 4 + (size_t)BB * HH * 4 + (size_t)BB * HH * 2;
  lstm_rec<<<dim3(2), 1024, lds, stream>>>(whh_bf, pre, seq);

  // emissions
  fc_kernel<<<MM, 32, 0, stream>>>(seq, Wfc, bfc, em);

  // CRF
  init_loss<<<1, 1, 0, stream>>>(out);
  crf_kernel<<<BB, 32, 0, stream>>>(em, mask, labels, startT, endT, trans, out,
                                    bp, 1.0f / (float)BB);
}
